// HGCNConv1_10969346474536
// MI455X (gfx1250) — matrix operand order repaired and checked by reading, not compiled, not run on hardware
//
#include <hip/hip_runtime.h>
#include <hip/hip_bf16.h>
#include <math.h>

typedef __attribute__((ext_vector_type(16))) __bf16 v16bf;
typedef __attribute__((ext_vector_type(8)))  __bf16 v8bf;
typedef __attribute__((ext_vector_type(8)))  float  v8f;

#define MIN_NORM 1e-15f
#define BALL_EPS 4e-3f

__device__ __forceinline__ float artanh_clip(float v) {
    v = fminf(fmaxf(v, -1.0f + 1e-7f), 1.0f - 1e-7f);
    return 0.5f * logf((1.0f + v) / (1.0f - v));
}

// Block-wide row reduction; blockDim.x must be a multiple of 32 (<=256).
__device__ __forceinline__ float rowReduceSum(float v, float* sbuf) {
    for (int off = 16; off > 0; off >>= 1) v += __shfl_xor(v, off, 32);
    const int nw  = blockDim.x >> 5;
    const int wid = threadIdx.x >> 5;
    if ((threadIdx.x & 31) == 0) sbuf[wid] = v;
    __syncthreads();
    float r = 0.f;
    for (int i = 0; i < nw; ++i) r += sbuf[i];
    __syncthreads();
    return r;
}

// ---------------------------------------------------------------------------
// Weight prep: Th/Tl = bf16 hi/lo split of proj(expmap0(W, c), c), W is [N,K]
// (already B^T layout for the GEMM). One block per row.
// ---------------------------------------------------------------------------
__global__ __launch_bounds__(256) void hgcn_prep_weight(
    const float* __restrict__ W, const float* __restrict__ cptr,
    __bf16* __restrict__ Th, __bf16* __restrict__ Tl, int K)
{
    __shared__ float sbuf[8];
    const size_t row = blockIdx.x;
    const float* w = W + row * (size_t)K;
    float s = 0.f;
    for (int k = threadIdx.x; k < K; k += blockDim.x) { float v = w[k]; s += v * v; }
    for (int off = 16; off > 0; off >>= 1) s += __shfl_xor(s, off, 32);
    if ((threadIdx.x & 31) == 0) sbuf[threadIdx.x >> 5] = s;
    __syncthreads();
    float tot = 0.f;
    const int nw = blockDim.x >> 5;
    for (int i = 0; i < nw; ++i) tot += sbuf[i];

    const float c = cptr[0], sc = sqrtf(c);
    const float un   = fmaxf(sqrtf(tot), MIN_NORM);
    const float s1   = tanhf(sc * un) / (sc * un);   // expmap0 scale
    const float nrm  = s1 * un;                      // ||expmap0(row)||
    const float maxn = (1.0f - BALL_EPS) / sc;
    const float scale = s1 * (nrm > maxn ? maxn / nrm : 1.0f);
    for (int k = threadIdx.x; k < K; k += blockDim.x) {
        float v  = w[k] * scale;
        __bf16 h = (__bf16)v;
        Th[row * (size_t)K + k] = h;
        Tl[row * (size_t)K + k] = (__bf16)(v - (float)h);
    }
}

// hb = proj(expmap0(bias, c), c), fp32 out. One block, OUT threads.
__global__ void hgcn_prep_bias(const float* __restrict__ b,
                               const float* __restrict__ cptr,
                               float* __restrict__ hb, int K)
{
    __shared__ float sbuf[8];
    const float v   = b[threadIdx.x];
    const float tot = rowReduceSum(v * v, sbuf);
    const float c = cptr[0], sc = sqrtf(c);
    const float un   = fmaxf(sqrtf(tot), MIN_NORM);
    const float s1   = tanhf(sc * un) / (sc * un);
    const float nrm  = s1 * un;
    const float maxn = (1.0f - BALL_EPS) / sc;
    hb[threadIdx.x] = v * s1 * (nrm > maxn ? maxn / nrm : 1.0f);
}

// ---------------------------------------------------------------------------
// WMMA GEMM: C[M,N] = A[M,K] (fp32) @ Bt^T where Bt is [N,K] bf16 hi/lo split.
// Emulated-fp32 via 3 bf16 WMMAs per K=32 step per 16x16 tile.
//
// Tiling: 256 threads = 8 waves; block covers 32 rows x 128 cols.
//   wave = mt*4 + ng :  mt in {0,1} -> m-tile,  ng in {0..3} -> 32-col group.
// Each wave converts ONE A fragment per k-step and feeds SIX WMMAs (two
// 16x16 accumulators), so fp32->bf16-hi/lo conversion VALU is amortized 2x
// and cross-wave redundancy is 4x (was 8x). Per-row sum-of-squares of A
// (needed by every hyperbolic tail) is computed only by ng==0 waves.
// Grid = M/32.  Requires N == 128, K % 32 == 0.
// ---------------------------------------------------------------------------
__global__ __launch_bounds__(256) void hgcn_gemm_wmma(
    const float* __restrict__ A, const __bf16* __restrict__ Bth,
    const __bf16* __restrict__ Btl, float* __restrict__ C,
    float* __restrict__ rowsq, int M, int N, int K)
{
    const int lane = threadIdx.x & 31;
    const int wave = threadIdx.x >> 5;
    const int mt   = wave >> 2;             // 0..1  m-tile within block
    const int ng   = wave & 3;              // 0..3  32-col group
    const int lh   = lane >> 4;             // lane half (K-space split)
    const int l16  = lane & 15;
    const size_t m0 = (size_t)blockIdx.x * 32 + mt * 16;
    const int    n0 = ng * 32;

    const float*  __restrict__ arow = A   + (m0 + l16) * (size_t)K;
    const __bf16* __restrict__ bh0r = Bth + (size_t)(n0      + l16) * K;
    const __bf16* __restrict__ bh1r = Bth + (size_t)(n0 + 16 + l16) * K;
    const __bf16* __restrict__ bl0r = Btl + (size_t)(n0      + l16) * K;
    const __bf16* __restrict__ bl1r = Btl + (size_t)(n0 + 16 + l16) * K;

    v8f acc0 = {0.f, 0.f, 0.f, 0.f, 0.f, 0.f, 0.f, 0.f};
    v8f acc1 = {0.f, 0.f, 0.f, 0.f, 0.f, 0.f, 0.f, 0.f};
    float sq = 0.f;

    for (int k0 = 0; k0 < K; k0 += 32) {
        // A layout (16-bit A 16x32): lane<16 -> K {0..7, 16..23}; lane>=16 -> +8
        const int a0 = k0 + lh * 8;
        float4 f0 = *(const float4*)(arow + a0);
        float4 f1 = *(const float4*)(arow + a0 + 4);
        float4 f2 = *(const float4*)(arow + a0 + 16);
        float4 f3 = *(const float4*)(arow + a0 + 20);
        if (k0 + 32 < K) __builtin_prefetch(arow + a0 + 32, 0, 3);

        const float av[16] = {f0.x, f0.y, f0.z, f0.w, f1.x, f1.y, f1.z, f1.w,
                              f2.x, f2.y, f2.z, f2.w, f3.x, f3.y, f3.z, f3.w};
        v16bf ah, al;
#pragma unroll
        for (int i = 0; i < 16; ++i) {
            float v = av[i];
            __bf16 h = (__bf16)v;
            ah[i] = h;
            al[i] = (__bf16)(v - (float)h);
        }
        if (ng == 0) {      // wave-uniform: only norm-producing waves pay this
#pragma unroll
            for (int i = 0; i < 16; ++i) sq += av[i] * av[i];
        }

        // B layout (32x16): lane holds one col, K {0..7, 16..23}; lane>=16 -> +8 (matches A)
        const int b0 = k0 + lh * 8;
        v8bf h00 = *(const v8bf*)(bh0r + b0);
        v8bf h01 = *(const v8bf*)(bh0r + b0 + 16);
        v8bf h10 = *(const v8bf*)(bh1r + b0);
        v8bf h11 = *(const v8bf*)(bh1r + b0 + 16);
        v8bf l00 = *(const v8bf*)(bl0r + b0);
        v8bf l01 = *(const v8bf*)(bl0r + b0 + 16);
        v8bf l10 = *(const v8bf*)(bl1r + b0);
        v8bf l11 = *(const v8bf*)(bl1r + b0 + 16);
        v16bf bh0 = __builtin_shufflevector(h00, h01, 0,1,2,3,4,5,6,7,8,9,10,11,12,13,14,15);
        v16bf bh1 = __builtin_shufflevector(h10, h11, 0,1,2,3,4,5,6,7,8,9,10,11,12,13,14,15);
        v16bf bl0 = __builtin_shufflevector(l00, l01, 0,1,2,3,4,5,6,7,8,9,10,11,12,13,14,15);
        v16bf bl1 = __builtin_shufflevector(l10, l11, 0,1,2,3,4,5,6,7,8,9,10,11,12,13,14,15);

        // Interleave the two accumulators to avoid back-to-back RAW on acc.
        acc0 = __builtin_amdgcn_wmma_f32_16x16x32_bf16(false, ah, false, bh0, (short)0, acc0, false, false);
        acc1 = __builtin_amdgcn_wmma_f32_16x16x32_bf16(false, ah, false, bh1, (short)0, acc1, false, false);
        acc0 = __builtin_amdgcn_wmma_f32_16x16x32_bf16(false, al, false, bh0, (short)0, acc0, false, false);
        acc1 = __builtin_amdgcn_wmma_f32_16x16x32_bf16(false, al, false, bh1, (short)0, acc1, false, false);
        acc0 = __builtin_amdgcn_wmma_f32_16x16x32_bf16(false, ah, false, bl0, (short)0, acc0, false, false);
        acc1 = __builtin_amdgcn_wmma_f32_16x16x32_bf16(false, ah, false, bl1, (short)0, acc1, false, false);
    }

    // Row sum-of-squares: lanes L and L+16 jointly cover one row's K space.
    sq += __shfl_xor(sq, 16, 32);
    if (ng == 0 && lh == 0 && rowsq) rowsq[m0 + l16] = sq;

    // C/D layout: VGPR r -> row m0 + r + 8*lh, col n0 + (lane&15)
#pragma unroll
    for (int r = 0; r < 8; ++r) {
        C[(m0 + 8 * lh + r) * (size_t)N + n0 +      l16] = acc0[r];
        C[(m0 + 8 * lh + r) * (size_t)N + n0 + 16 + l16] = acc1[r];
    }
}

// ---------------------------------------------------------------------------
// hyp_linear tail: h1 = proj(mobius_add(proj(mobius_from_mx(mx, ||x||)), hb))
// One block (OUT threads) per row.
// ---------------------------------------------------------------------------
__global__ void hgcn_linear_tail(const float* __restrict__ MX,
    const float* __restrict__ xrowsq, const float* __restrict__ hb,
    const float* __restrict__ cptr, float* __restrict__ H, int OUT)
{
    __shared__ float sbuf[8];
    const int j = threadIdx.x;
    const size_t i = blockIdx.x;
    const float c = cptr[0], sc = sqrtf(c);
    const float maxn = (1.0f - BALL_EPS) / sc;

    float mx  = MX[i * OUT + j];
    float mx2 = rowReduceSum(mx * mx, sbuf);
    const bool zero = (mx2 == 0.0f);
    float mxn = fmaxf(sqrtf(mx2), MIN_NORM);
    float xn  = fmaxf(sqrtf(xrowsq[i]), MIN_NORM);
    float r   = tanhf(mxn / xn * artanh_clip(sc * xn)) / (mxn * sc);
    float res = zero ? 0.0f : mx * r;

    float rn = fmaxf(sqrtf(rowReduceSum(res * res, sbuf)), MIN_NORM);
    if (rn > maxn) res *= maxn / rn;

    const float y  = hb[j];
    float x2 = rowReduceSum(res * res, sbuf);
    float y2 = rowReduceSum(y * y, sbuf);
    float xy = rowReduceSum(res * y, sbuf);
    float den = fmaxf(1.0f + 2.0f * c * xy + c * c * x2 * y2, MIN_NORM);
    float o = ((1.0f + 2.0f * c * xy + c * y2) * res + (1.0f - c * x2) * y) / den;

    float on = fmaxf(sqrtf(rowReduceSum(o * o, sbuf)), MIN_NORM);
    if (on > maxn) o *= maxn / on;
    H[i * OUT + j] = o;
}

// support = mobius_from_mx(mx2, ||h1||)
__global__ void hgcn_support_tail(const float* __restrict__ MX,
    const float* __restrict__ hrowsq, const float* __restrict__ cptr,
    float* __restrict__ S, int OUT)
{
    __shared__ float sbuf[8];
    const int j = threadIdx.x;
    const size_t i = blockIdx.x;
    const float c = cptr[0], sc = sqrtf(c);
    float mx  = MX[i * OUT + j];
    float mx2 = rowReduceSum(mx * mx, sbuf);
    const bool zero = (mx2 == 0.0f);
    float mxn = fmaxf(sqrtf(mx2), MIN_NORM);
    float hn  = fmaxf(sqrtf(hrowsq[i]), MIN_NORM);
    float r   = tanhf(mxn / hn * artanh_clip(sc * hn)) / (mxn * sc);
    S[i * OUT + j] = zero ? 0.0f : mx * r;
}

// support [R,C] fp32 -> St [C,R] bf16 hi/lo (B^T layout for GEMM3)
__global__ __launch_bounds__(256) void hgcn_transpose_split(
    const float* __restrict__ S, __bf16* __restrict__ Th,
    __bf16* __restrict__ Tl, int R, int Ccols)
{
    __shared__ float tile[32][33];
    const int r0 = blockIdx.x * 32, c0 = blockIdx.y * 32;
    const int tx = threadIdx.x, ty = threadIdx.y;   // 32 x 8
#pragma unroll
    for (int j = 0; j < 4; ++j)
        tile[ty + 8 * j][tx] = S[(size_t)(r0 + ty + 8 * j) * Ccols + c0 + tx];
    __syncthreads();
#pragma unroll
    for (int j = 0; j < 4; ++j) {
        float v  = tile[tx][ty + 8 * j];
        __bf16 h = (__bf16)v;
        size_t o = (size_t)(c0 + ty + 8 * j) * R + r0 + tx;
        Th[o] = h;
        Tl[o] = (__bf16)(v - (float)h);
    }
}

// hyp_agg tail (mobius rescale by ||adj row||, raw-bias mobius_add, proj)
// fused with hyp_act (logmap0 -> leaky_relu -> expmap0(1-c) -> proj(1-c)).
__global__ void hgcn_agg_act_tail(const float* __restrict__ MX,
    const float* __restrict__ arowsq, const float* __restrict__ bias,
    const float* __restrict__ cptr, float* __restrict__ Y, int OUT)
{
    __shared__ float sbuf[8];
    const int j = threadIdx.x;
    const size_t i = blockIdx.x;
    const float c = cptr[0], sc = sqrtf(c);
    const float maxn = (1.0f - BALL_EPS) / sc;

    float mx  = MX[i * OUT + j];
    float mx2 = rowReduceSum(mx * mx, sbuf);
    const bool zero = (mx2 == 0.0f);
    float mxn = fmaxf(sqrtf(mx2), MIN_NORM);
    float an  = fmaxf(sqrtf(arowsq[i]), MIN_NORM);
    float r   = tanhf(mxn / an * artanh_clip(sc * an)) / (mxn * sc);
    float res = zero ? 0.0f : mx * r;

    // mobius_add(res, bias) -- raw Euclidean bias, no pre-proj (matches source)
    const float y = bias[j];
    float x2 = rowReduceSum(res * res, sbuf);
    float y2 = rowReduceSum(y * y, sbuf);
    float xy = rowReduceSum(res * y, sbuf);
    float den = fmaxf(1.0f + 2.0f * c * xy + c * c * x2 * y2, MIN_NORM);
    float o = ((1.0f + 2.0f * c * xy + c * y2) * res + (1.0f - c * x2) * y) / den;

    float on = fmaxf(sqrtf(rowReduceSum(o * o, sbuf)), MIN_NORM);
    if (on > maxn) o *= maxn / on;

    // hyp_act: logmap0 with c, leaky_relu, expmap0/proj with c2 = 1 - c
    float pn = fmaxf(sqrtf(rowReduceSum(o * o, sbuf)), MIN_NORM);
    float xt = artanh_clip(sc * pn) * o / (pn * sc);
    xt = xt > 0.0f ? xt : 0.01f * xt;

    const float c2 = 1.0f - c, sc2 = sqrtf(c2);
    float un = fmaxf(sqrtf(rowReduceSum(xt * xt, sbuf)), MIN_NORM);
    float e  = tanhf(sc2 * un) * xt / (sc2 * un);
    float en = fmaxf(sqrtf(rowReduceSum(e * e, sbuf)), MIN_NORM);
    const float maxn2 = (1.0f - BALL_EPS) / sc2;
    if (en > maxn2) e *= maxn2 / en;
    Y[i * OUT + j] = e;
}

extern "C" void kernel_launch(void* const* d_in, const int* in_sizes, int n_in,
                              void* d_out, int out_size, void* d_ws, size_t ws_size,
                              hipStream_t stream) {
    const float* x   = (const float*)d_in[0];
    const float* adj = (const float*)d_in[1];
    const float* cp  = (const float*)d_in[2];
    const float* lw  = (const float*)d_in[3];
    const float* lb  = (const float*)d_in[4];
    const float* aw  = (const float*)d_in[5];
    const float* ab  = (const float*)d_in[6];

    const int OUT = in_sizes[4];          // 128
    const int IN  = in_sizes[3] / OUT;    // 512
    const int M   = in_sizes[0] / IN;     // 8192

    char* base = (char*)d_ws;
    size_t off = 0;
    auto carve = [&](size_t bytes) -> void* {
        void* p = base + off;
        off += (bytes + 255) & ~(size_t)255;
        return p;
    };
    __bf16* hw1h = (__bf16*)carve((size_t)OUT * IN  * sizeof(__bf16));
    __bf16* hw1l = (__bf16*)carve((size_t)OUT * IN  * sizeof(__bf16));
    __bf16* hw2h = (__bf16*)carve((size_t)OUT * OUT * sizeof(__bf16));
    __bf16* hw2l = (__bf16*)carve((size_t)OUT * OUT * sizeof(__bf16));
    float*  hb   = (float*)carve((size_t)OUT * sizeof(float));
    float*  mx   = (float*)carve((size_t)M * OUT * sizeof(float));
    float*  h1   = (float*)carve((size_t)M * OUT * sizeof(float));
    float*  sup  = (float*)carve((size_t)M * OUT * sizeof(float));
    __bf16* sth  = (__bf16*)carve((size_t)OUT * M * sizeof(__bf16));
    __bf16* stl  = (__bf16*)carve((size_t)OUT * M * sizeof(__bf16));
    float*  xsq  = (float*)carve((size_t)M * sizeof(float));
    float*  hsq  = (float*)carve((size_t)M * sizeof(float));
    float*  asq  = (float*)carve((size_t)M * sizeof(float));

    // Weight / bias prep (hyperbolic map of parameters)
    hgcn_prep_weight<<<OUT, 256, 0, stream>>>(lw, cp, hw1h, hw1l, IN);
    hgcn_prep_weight<<<OUT, 256, 0, stream>>>(aw, cp, hw2h, hw2l, OUT);
    hgcn_prep_bias<<<1, OUT, 0, stream>>>(lb, cp, hb, OUT);

    // hyp_linear: GEMM1 (also produces ||x row||^2) + tail
    hgcn_gemm_wmma<<<M / 32, 256, 0, stream>>>(x, hw1h, hw1l, mx, xsq, M, OUT, IN);
    hgcn_linear_tail<<<M, OUT, 0, stream>>>(mx, xsq, hb, cp, h1, OUT);

    // hyp_agg part 1: GEMM2 (also ||h1 row||^2) + support tail
    hgcn_gemm_wmma<<<M / 32, 256, 0, stream>>>(h1, hw2h, hw2l, mx, hsq, M, OUT, OUT);
    hgcn_support_tail<<<M, OUT, 0, stream>>>(mx, hsq, cp, sup, OUT);

    // hyp_agg part 2: transpose support to B^T bf16 hi/lo, big adj GEMM
    hgcn_transpose_split<<<dim3(M / 32, OUT / 32), dim3(32, 8), 0, stream>>>(sup, sth, stl, M, OUT);
    hgcn_gemm_wmma<<<M / 32, 256, 0, stream>>>(adj, sth, stl, mx, asq, M, OUT, M);

    // hyp_agg tail + hyp_act fused -> final output
    hgcn_agg_act_tail<<<M, OUT, 0, stream>>>(mx, asq, ab, cp, (float*)d_out, OUT);
}
